// AttractorDynamics_68229850464922
// MI455X (gfx1250) — matrix operand order repaired and checked by reading, not compiled, not run on hardware
//
#include <hip/hip_runtime.h>
#include <stdint.h>

// ---- problem constants (match reference) ----
#define BATCH 16384
#define MDIM  1024
#define ADIM  512
#define STEPS 10
#define TAU   0.1f

#define TM     64       // sigma rows owned by one workgroup
#define NTILES 16       // 16x16 n-tiles per wave (256 columns)
#define BLOCK  256      // 8 waves of 32
#define NCH1   (MDIM/32)   // 32 k-chunks, phase 1
#define NCH2   (ADIM/32)   // 16 k-chunks, phase 2

// LDS strides padded for bank-conflict avoidance + 16B fragment alignment
#define BT_STRIDE  20    // uint32 (bf16 pair) per B row: 16 pairs + pad, %4==0
#define AT_STRIDE  20
#define SIG_STRIDE 520   // uint16 per sigma row (byte stride 1040 %16==0)

#define DRVF_BYTES (BLOCK * NTILES * 8 * 4)  // frag-major drive: 131072
#define BT_BYTES   (ADIM * BT_STRIDE * 4)    // 40960 (one buffer)
#define AT_BYTES   (TM * AT_STRIDE * 4)      // 5120  (one buffer)
#define SIG_BYTES  (TM * SIG_STRIDE * 2)     // 66560
#define BT0_OFF    DRVF_BYTES
#define AT0_OFF    (DRVF_BYTES + 2*BT_BYTES)
#define SIG_OFF    (DRVF_BYTES + 2*BT_BYTES + 2*AT_BYTES)
#define SMEM_BYTES (SIG_OFF + SIG_BYTES)     // 289792

#define CHUNK_BYTES (ADIM * 64)              // 512 rows * 16 bf16-pairs = 32 KB

typedef __attribute__((ext_vector_type(16))) __bf16   v16bf;
typedef __attribute__((ext_vector_type(8)))  float    v8f;
typedef __attribute__((ext_vector_type(8)))  uint32_t v8u;

static __device__ __forceinline__ uint32_t f2bf(float f) {
  uint32_t u = __float_as_uint(f);
  u += 0x7FFFu + ((u >> 16) & 1u);           // round-to-nearest-even
  return u >> 16;
}
static __device__ __forceinline__ uint32_t pack2(float lo, float hi) {
  return f2bf(lo) | (f2bf(hi) << 16);
}
static __device__ __forceinline__ float fast_tanh(float x) {
#if __has_builtin(__builtin_amdgcn_tanhf)
  return __builtin_amdgcn_tanhf(x);          // V_TANH_F32 (1 TRANS op)
#else
  return tanhf(x);
#endif
}
// two aligned 16B LDS reads -> one bf16 WMMA fragment
static __device__ __forceinline__ v16bf ld_frag(const uint32_t* p0, const uint32_t* p1) {
  uint4 a = *(const uint4*)p0;               // ds_load_b128
  uint4 b = *(const uint4*)p1;               // ds_load_b128
  v8u u = {a.x, a.y, a.z, a.w, b.x, b.y, b.z, b.w};
  return __builtin_bit_cast(v16bf, u);
}
// CDNA5 async global->LDS copy (ASYNCcnt path, no VGPR data, no VALU convert)
static __device__ __forceinline__ void async_cp16(uint32_t lds_off, const void* gptr) {
  asm volatile("global_load_async_to_lds_b128 %0, %1, off"
               :: "v"(lds_off), "v"((uint64_t)(uintptr_t)gptr) : "memory");
}
#define WAIT_ASYNC0() asm volatile("s_wait_asynccnt 0x0" ::: "memory")

// Pin the chunk-body schedule: 8 DS reads up front (A-frag + 3 B-frags),
// then 1 WMMA : 2 DS reads interleave, then 3 tail WMMAs. Forces fragment
// loads >=3 tiles ahead of their consumers so s_wait_dscnt relaxes.
// mask 0x100 = DS-read group, mask 0x008 = WMMA group. Best-effort hint.
static __device__ __forceinline__ void sched_chunk_hint() {
#if __has_builtin(__builtin_amdgcn_sched_group_barrier)
  __builtin_amdgcn_sched_group_barrier(0x100, 8, 0);
#pragma unroll
  for (int s = 0; s < 13; ++s) {
    __builtin_amdgcn_sched_group_barrier(0x008, 1, 0);
    __builtin_amdgcn_sched_group_barrier(0x100, 2, 0);
  }
  __builtin_amdgcn_sched_group_barrier(0x008, 3, 0);
#endif
}

// issue one 32KB B-operand chunk (512 rows x 64B) into a padded LDS buffer
static __device__ __forceinline__ void issue_bchunk(const char* src, uint32_t dst_off,
                                                    int tid) {
#pragma unroll
  for (int i = tid; i < ADIM * 4; i += BLOCK) {        // 8 issues / thread
    const int n = i >> 2, s = i & 3;
    async_cp16(dst_off + (uint32_t)(n * (BT_STRIDE * 4) + s * 16), src + i * 16);
  }
}

// ---------------- prep: bf16-pack W_in and (J * 1/tau), chunk-major ----------
__global__ void __launch_bounds__(BLOCK)
prep_bf16_kernel(const float* __restrict__ Win, const float* __restrict__ Jm,
                 uint32_t* __restrict__ Wbf, uint32_t* __restrict__ Jbf) {
  const int i = blockIdx.x * BLOCK + threadIdx.x;
  const float inv_tau = 1.0f / TAU;
  if (i < NCH2 * ADIM * 16) {                 // J: 16 chunks x 512 rows x 16 pairs
    const int c = i & 15, n = (i >> 4) & (ADIM - 1), kc = i >> 13;
    const float2 v = *(const float2*)(Jm + (size_t)n * ADIM + kc * 32 + 2 * c);
    Jbf[i] = pack2(v.x * inv_tau, v.y * inv_tau);   // fold 1/tau into J
  }
  if (i < NCH1 * ADIM * 16) {                 // W: 32 chunks x 512 rows x 16 pairs
    const int c = i & 15, n = (i >> 4) & (ADIM - 1), kc = i >> 13;
    const float2 v = *(const float2*)(Win + (size_t)n * MDIM + kc * 32 + 2 * c);
    Wbf[i] = pack2(v.x, v.y);
  }
}

// ---------------- fused attractor kernel ------------------------------------
__global__ void __launch_bounds__(BLOCK, 1)
attractor_fused_kernel(const float* __restrict__ x,
                       const float* __restrict__ bin,
                       const char*  __restrict__ wbf,   // bf16 W, chunk-major
                       const char*  __restrict__ jbf,   // bf16 J/tau, chunk-major
                       float* __restrict__ out) {
  extern __shared__ char smem[];
  // LDS-derived pointers stay scalar locals (no aggregate initializers).
  float*    driveF = (float*)smem;                 // frag-major, wave-private
  uint16_t* sigmaS = (uint16_t*)(smem + SIG_OFF);

  const int tid  = threadIdx.x;
  const int lane = tid & 31;
  const int wv   = tid >> 5;
  const int wr   = wv & 3;          // row-tile 0..3
  const int wc   = wv >> 2;         // column half 0..1
  const int ln   = lane & 15;
  const int kh   = lane >> 4;
  const int row0 = blockIdx.x * TM;

  auto bTile  = [&](int b) -> uint32_t* {
    return (uint32_t*)(smem + BT0_OFF + b * BT_BYTES);
  };
  auto aTile  = [&](int b) -> uint32_t* {
    return (uint32_t*)(smem + AT0_OFF + b * AT_BYTES);
  };
  auto bt_off = [&](int b) -> uint32_t {
    return (uint32_t)(uintptr_t)(smem + BT0_OFF + b * BT_BYTES);
  };
  // per-lane contiguous slot for this wave's drive fragment `nt`
  auto dslot = [&](int nt) -> float* {
    return driveF + (((wv * NTILES + nt) * 32 + lane) << 3);
  };

  v8f acc[NTILES];
#pragma unroll
  for (int nt = 0; nt < NTILES; ++nt) acc[nt] = (v8f){0,0,0,0,0,0,0,0};

  // stage one x chunk (64x32 f32 -> bf16 pairs) into aTile(b)
  auto stage_x = [&](int kc, int b) {
    uint32_t* at = aTile(b);
#pragma unroll
    for (int i = tid; i < TM * 16; i += BLOCK) {       // 4 iters
      const int m = i >> 4, c = i & 15;
      const float2 v = *(const float2*)(x + (size_t)(row0 + m) * MDIM + kc * 32 + 2 * c);
      at[m * AT_STRIDE + c] = pack2(v.x, v.y);
    }
  };

  // ============ Phase 1: drive = x @ Win^T, async-pipelined ============
  issue_bchunk(wbf, bt_off(0), tid);                   // prologue: W chunk 0
  stage_x(0, 0);
#pragma unroll 1
  for (int kc = 0; kc < NCH1; ++kc) {
    const int b = kc & 1;
    WAIT_ASYNC0();                                     // my chunk-kc copies done
    __syncthreads();                                   // all staged; prev reads done
    if (kc + 1 < NCH1) {                               // overlap copy of kc+1
      issue_bchunk(wbf + (size_t)(kc + 1) * CHUNK_BYTES, bt_off((kc + 1) & 1), tid);
      stage_x(kc + 1, (kc + 1) & 1);
    }
    const uint32_t* ap = aTile(b) + (16 * wr + ln) * AT_STRIDE + kh * 4;
    const v16bf afrag  = ld_frag(ap, ap + 8);
    const uint32_t* bb = bTile(b) + ln * BT_STRIDE + kh * 4;
    auto ldb = [&](int nt) -> v16bf {
      const uint32_t* bp = bb + (256 * wc + 16 * nt) * BT_STRIDE;
      return ld_frag(bp, bp + 8);
    };
    // depth-3 rotation + pinned schedule below
    v16bf bf0 = ldb(0);
    v16bf bf1 = ldb(1);
    v16bf bf2 = ldb(2);
#pragma unroll
    for (int nt = 0; nt < NTILES; ++nt) {
      v16bf bnew = (nt + 3 < NTILES) ? ldb(nt + 3) : bf0;
      acc[nt] = __builtin_amdgcn_wmma_f32_16x16x32_bf16(
          false, afrag, false, bf0, (short)0, acc[nt], false, false);
      bf0 = bf1; bf1 = bf2; bf2 = bnew;
    }
    sched_chunk_hint();
  }

  // bias, park drive in LDS (frag-major b128), sigma_1 = tanh(drive)
#pragma unroll
  for (int nt = 0; nt < NTILES; ++nt) {
    const int ncol = 256 * wc + 16 * nt + ln;
    const float bias = bin[ncol];
    float d[8];
#pragma unroll
    for (int v = 0; v < 8; ++v) d[v] = acc[nt][v] + bias;
    float* p = dslot(nt);
    *(float4*)(p)     = make_float4(d[0], d[1], d[2], d[3]);  // ds_store_b128
    *(float4*)(p + 4) = make_float4(d[4], d[5], d[6], d[7]);  // ds_store_b128
#pragma unroll
    for (int v = 0; v < 8; ++v)
      sigmaS[(16 * wr + 8 * kh + v) * SIG_STRIDE + ncol] = (uint16_t)f2bf(fast_tanh(d[v]));
  }

  // ============ Phase 2: 9 more settling steps ============
#pragma unroll 1
  for (int it = 1; it < STEPS; ++it) {
    // C := drive (wave-private slots; two b128 loads per tile)
#pragma unroll
    for (int nt = 0; nt < NTILES; ++nt) {
      const float* p = dslot(nt);
      const float4 lo = *(const float4*)(p);
      const float4 hi = *(const float4*)(p + 4);
      acc[nt] = (v8f){lo.x, lo.y, lo.z, lo.w, hi.x, hi.y, hi.z, hi.w};
    }
    issue_bchunk(jbf, bt_off(0), tid);                 // prologue: J chunk 0

#pragma unroll 1
    for (int kc = 0; kc < NCH2; ++kc) {
      const int b = kc & 1;
      WAIT_ASYNC0();
      __syncthreads();
      if (kc + 1 < NCH2)
        issue_bchunk(jbf + (size_t)(kc + 1) * CHUNK_BYTES, bt_off((kc + 1) & 1), tid);

      const uint16_t* sp = sigmaS + (16 * wr + ln) * SIG_STRIDE + kc * 32 + kh * 8;
      const v16bf afrag  = ld_frag((const uint32_t*)sp, (const uint32_t*)(sp + 16));
      const uint32_t* bb = bTile(b) + ln * BT_STRIDE + kh * 4;
      auto ldb = [&](int nt) -> v16bf {
        const uint32_t* bp = bb + (256 * wc + 16 * nt) * BT_STRIDE;
        return ld_frag(bp, bp + 8);
      };
      v16bf bf0 = ldb(0);
      v16bf bf1 = ldb(1);
      v16bf bf2 = ldb(2);
#pragma unroll
      for (int nt = 0; nt < NTILES; ++nt) {
        v16bf bnew = (nt + 3 < NTILES) ? ldb(nt + 3) : bf0;
        acc[nt] = __builtin_amdgcn_wmma_f32_16x16x32_bf16(
            false, afrag, false, bf0, (short)0, acc[nt], false, false);
        bf0 = bf1; bf1 = bf2; bf2 = bnew;
      }
      sched_chunk_hint();
    }

    __syncthreads();            // all sigma reads done before overwrite
    const bool last = (it == STEPS - 1);
#pragma unroll
    for (int nt = 0; nt < NTILES; ++nt) {
      const int ncol = 256 * wc + 16 * nt + ln;
#pragma unroll
      for (int v = 0; v < 8; ++v) {
        const int r = 16 * wr + 8 * kh + v;
        const float s = fast_tanh(acc[nt][v]);
        if (last)
          out[(size_t)(row0 + r) * ADIM + ncol] = s;   // coalesced f32 store
        else
          sigmaS[r * SIG_STRIDE + ncol] = (uint16_t)f2bf(s);
      }
    }
  }
}

extern "C" void kernel_launch(void* const* d_in, const int* in_sizes, int n_in,
                              void* d_out, int out_size, void* d_ws, size_t ws_size,
                              hipStream_t stream) {
  (void)in_sizes; (void)n_in; (void)out_size; (void)ws_size;
  const float* x   = (const float*)d_in[0];
  const float* Win = (const float*)d_in[1];
  const float* bin = (const float*)d_in[2];
  const float* Jm  = (const float*)d_in[3];
  float* out = (float*)d_out;

  // workspace layout: [Jbf: 512KB][Wbf: 1MB]
  uint32_t* Jbf = (uint32_t*)d_ws;
  uint32_t* Wbf = (uint32_t*)((char*)d_ws + (size_t)NCH2 * CHUNK_BYTES);

  prep_bf16_kernel<<<dim3((NCH1 * ADIM * 16 + BLOCK - 1) / BLOCK), dim3(BLOCK),
                     0, stream>>>(Win, Jm, Wbf, Jbf);

  (void)hipFuncSetAttribute((const void*)attractor_fused_kernel,
                            hipFuncAttributeMaxDynamicSharedMemorySize,
                            (int)SMEM_BYTES);
  attractor_fused_kernel<<<dim3(BATCH / TM), dim3(BLOCK), SMEM_BYTES, stream>>>(
      x, bin, (const char*)Wbf, (const char*)Jbf, out);
}
